// GCN_71631464562988
// MI455X (gfx1250) — matrix operand order repaired and checked by reading, not compiled, not run on hardware
//
#include <hip/hip_runtime.h>
#include <hip/hip_bf16.h>

// ---------------------------------------------------------------------------
// GCN inference for MI455X (gfx1250, wave32, WMMA).
//   out = softmax( A @ ( relu( A @ (X@W1) ) @ W2 ) )
// GEMMs via v_wmma_f32_16x16x32_bf16; sparse propagation via coalesced
// gather + global f32 atomic scatter (dest buffers fit in 192MB L2).
// ---------------------------------------------------------------------------

#define N_NODES     50000
#define N_EDGES     1600000
#define D_FEAT      512
#define FILTERS     256
#define NUM_CLASSES 16

typedef __attribute__((ext_vector_type(16))) __bf16 v16bf;
typedef __attribute__((ext_vector_type(8)))  float  v8f;

__device__ __forceinline__ unsigned short f32_to_bf16_rne(float f) {
  union { float f; unsigned u; } v; v.f = f;
  unsigned r = v.u + 0x7FFFu + ((v.u >> 16) & 1u);   // round to nearest even
  return (unsigned short)(r >> 16);
}

// ---- elementwise f32 -> bf16 (optionally fused relu), 4 elems/thread ------
__global__ void convert_f32_bf16(const float* __restrict__ in,
                                 unsigned short* __restrict__ out,
                                 long n4, int relu) {
  long i = (long)blockIdx.x * blockDim.x + threadIdx.x;
  if (i >= n4) return;
  float4 f = ((const float4*)in)[i];
  if (relu) {
    f.x = fmaxf(f.x, 0.f); f.y = fmaxf(f.y, 0.f);
    f.z = fmaxf(f.z, 0.f); f.w = fmaxf(f.w, 0.f);
  }
  ushort4 o;
  o.x = f32_to_bf16_rne(f.x); o.y = f32_to_bf16_rne(f.y);
  o.z = f32_to_bf16_rne(f.z); o.w = f32_to_bf16_rne(f.w);
  ((ushort4*)out)[i] = o;
}

// ---- W [K x N] f32 row-major  ->  Wt [N x K] bf16 row-major ---------------
__global__ void transpose_f32_bf16(const float* __restrict__ in,
                                   unsigned short* __restrict__ out,
                                   int K, int N) {
  int idx = blockIdx.x * blockDim.x + threadIdx.x;
  if (idx >= K * N) return;
  int k = idx / N, n = idx % N;
  out[(size_t)n * K + k] = f32_to_bf16_rne(in[idx]);
}

// ---- WMMA GEMM:  C[MxN] f32 = A[MxK] bf16  @  Bt[NxK]^T bf16 --------------
// One wave computes one 16x16 tile. A/B frags use contiguous k order (16*half + s); C/D per CDNA5 ISA 7.12.2.
template <int K, int N>
__global__ void gemm_bf16_wmma(const unsigned short* __restrict__ A,
                               const unsigned short* __restrict__ Bt,
                               float* __restrict__ C, int M) {
  const int wave = (blockIdx.x * blockDim.x + threadIdx.x) >> 5;
  const int lane = threadIdx.x & 31;
  constexpr int NT = N >> 4;
  const int mt = wave / NT;
  const int nt = wave % NT;
  if (mt * 16 >= M) return;

  const int half = lane >> 4;       // 0: lanes 0-15, 1: lanes 16-31
  const int l    = lane & 15;

  const unsigned short* arow = A  + (size_t)(mt * 16 + l) * K;   // A row l
  const unsigned short* brow = Bt + (size_t)(nt * 16 + l) * K;   // B column l

  v8f acc = {};
#pragma unroll
  for (int k = 0; k < K; k += 32) {
    union { v16bf v; uint4 q[2]; } a, b;
    // A frag: elems 0..15 = K (k + 16*half .. +15), contiguous in A
    a.q[0] = *(const uint4*)(arow + k + half * 16);
    a.q[1] = *(const uint4*)(arow + k + half * 16 + 8);
    // B frag: elems 0..15 = K (k + 16*half .. +15), contiguous in Bt
    const uint4* bp = (const uint4*)(brow + k + half * 16);
    b.q[0] = bp[0];
    b.q[1] = bp[1];
    acc = __builtin_amdgcn_wmma_f32_16x16x32_bf16(
        /*neg_a=*/false, a.v, /*neg_b=*/false, b.v,
        /*c_mod=*/(short)0, acc, /*reuse_a=*/false, /*reuse_b=*/false);
  }
  // C/D layout: VGPR v, lane -> (M = v + 8*half, N = l)
  float* crow = C + (size_t)(mt * 16 + half * 8) * N + nt * 16 + l;
#pragma unroll
  for (int v = 0; v < 8; ++v) crow[(size_t)v * N] = acc[v];
}

// ---- propagation, 256 features: one wave per edge, 8 floats/lane ----------
__global__ void propagate_wide(const float* __restrict__ Hin,
                               const int* __restrict__ src,
                               const int* __restrict__ dst,
                               const float* __restrict__ w,
                               float* __restrict__ Hout, int nEdges) {
  int e    = (blockIdx.x * blockDim.x + threadIdx.x) >> 5;
  int lane = threadIdx.x & 31;
  if (e >= nEdges) return;
  int   s  = src[e];
  int   d  = dst[e];
  float ww = w[e];
  const float4* hi = (const float4*)(Hin + (size_t)s * FILTERS);
  float4 a = hi[lane];        // feats [lane*4 .. +3]
  float4 b = hi[lane + 32];   // feats [128 + lane*4 .. +3]
  float* o = Hout + (size_t)d * FILTERS;
  atomicAdd(o + lane * 4 + 0,       ww * a.x);
  atomicAdd(o + lane * 4 + 1,       ww * a.y);
  atomicAdd(o + lane * 4 + 2,       ww * a.z);
  atomicAdd(o + lane * 4 + 3,       ww * a.w);
  atomicAdd(o + 128 + lane * 4 + 0, ww * b.x);
  atomicAdd(o + 128 + lane * 4 + 1, ww * b.y);
  atomicAdd(o + 128 + lane * 4 + 2, ww * b.z);
  atomicAdd(o + 128 + lane * 4 + 3, ww * b.w);
}

// ---- propagation, 16 features: one thread per (edge, feature) -------------
__global__ void propagate_narrow(const float* __restrict__ Hin,
                                 const int* __restrict__ src,
                                 const int* __restrict__ dst,
                                 const float* __restrict__ w,
                                 float* __restrict__ out, int nEdges) {
  long gid = (long)blockIdx.x * blockDim.x + threadIdx.x;
  long e   = gid >> 4;
  int  f   = (int)(gid & 15);
  if (e >= nEdges) return;
  int   s  = src[e];
  int   d  = dst[e];
  float ww = w[e];
  atomicAdd(out + (size_t)d * NUM_CLASSES + f,
            ww * Hin[(size_t)s * NUM_CLASSES + f]);
}

// ---- softmax over 16 classes, one thread per node -------------------------
__global__ void softmax16(const float* __restrict__ logits,
                          float* __restrict__ out, int n) {
  int i = blockIdx.x * blockDim.x + threadIdx.x;
  if (i >= n) return;
  const float4* lp = (const float4*)(logits + (size_t)i * NUM_CLASSES);
  float4 q[4];
  q[0] = lp[0]; q[1] = lp[1]; q[2] = lp[2]; q[3] = lp[3];
  float* v = (float*)q;
  float m = v[0];
#pragma unroll
  for (int c = 1; c < 16; ++c) m = fmaxf(m, v[c]);
  float sum = 0.f;
#pragma unroll
  for (int c = 0; c < 16; ++c) { v[c] = __expf(v[c] - m); sum += v[c]; }
  float inv = 1.f / sum;
  float4* op = (float4*)(out + (size_t)i * NUM_CLASSES);
#pragma unroll
  for (int c = 0; c < 16; ++c) v[c] *= inv;
  op[0] = q[0]; op[1] = q[1]; op[2] = q[2]; op[3] = q[3];
}

// ---------------------------------------------------------------------------
extern "C" void kernel_launch(void* const* d_in, const int* in_sizes, int n_in,
                              void* d_out, int out_size, void* d_ws, size_t ws_size,
                              hipStream_t stream) {
  const float* x    = (const float*)d_in[0];
  const float* W1   = (const float*)d_in[1];
  const float* W2   = (const float*)d_in[2];
  const float* ew   = (const float*)d_in[3];
  const int*   esrc = (const int*)d_in[4];
  const int*   edst = (const int*)d_in[5];
  float* out = (float*)d_out;

  // Workspace layout (dead buffers aliased; ~154 MB total)
  char* base = (char*)d_ws;
  size_t off = 0;
  auto alloc = [&](size_t b) { size_t r = off; off += (b + 255) & ~(size_t)255; return r; };
  size_t xb_off  = alloc((size_t)N_NODES * D_FEAT * 2);        // Xb, later Hb
  size_t w1t_off = alloc((size_t)FILTERS * D_FEAT * 2);
  size_t w2t_off = alloc((size_t)NUM_CLASSES * FILTERS * 2);
  size_t xw1_off = alloc((size_t)N_NODES * FILTERS * 4);       // XW1, later logits
  size_t h_off   = alloc((size_t)N_NODES * FILTERS * 4);       // H, later HW2

  unsigned short* Xb     = (unsigned short*)(base + xb_off);
  unsigned short* W1t    = (unsigned short*)(base + w1t_off);
  unsigned short* W2t    = (unsigned short*)(base + w2t_off);
  float*          XW1    = (float*)(base + xw1_off);
  float*          H      = (float*)(base + h_off);
  unsigned short* Hb     = (unsigned short*)(base + xb_off);   // alias (Xb dead)
  float*          HW2    = (float*)(base + h_off);             // alias (H dead)
  float*          logits = (float*)(base + xw1_off);           // alias (XW1 dead)

  // 1) X -> bf16
  {
    long n4 = (long)N_NODES * D_FEAT / 4;
    convert_f32_bf16<<<(int)((n4 + 255) / 256), 256, 0, stream>>>(x, Xb, n4, 0);
  }
  // 2) W1, W2 -> bf16, transposed (column-major for contiguous B-frag loads)
  transpose_f32_bf16<<<(D_FEAT * FILTERS + 255) / 256, 256, 0, stream>>>(W1, W1t, D_FEAT, FILTERS);
  transpose_f32_bf16<<<(FILTERS * NUM_CLASSES + 255) / 256, 256, 0, stream>>>(W2, W2t, FILTERS, NUM_CLASSES);
  // 3) XW1 = X @ W1   (3125 x 16 tiles, 8 waves/block)
  {
    int waves = (N_NODES / 16) * (FILTERS / 16);
    gemm_bf16_wmma<D_FEAT, FILTERS><<<(waves + 7) / 8, 256, 0, stream>>>(Xb, W1t, XW1, N_NODES);
  }
  // 4) H = A @ XW1
  hipMemsetAsync(H, 0, (size_t)N_NODES * FILTERS * 4, stream);
  propagate_wide<<<N_EDGES / 8, 256, 0, stream>>>(XW1, esrc, edst, ew, H, N_EDGES);
  // 5) Hb = bf16(relu(H))
  {
    long n4 = (long)N_NODES * FILTERS / 4;
    convert_f32_bf16<<<(int)((n4 + 255) / 256), 256, 0, stream>>>(H, Hb, n4, 1);
  }
  // 6) HW2 = relu(H) @ W2
  {
    int waves = (N_NODES / 16) * (NUM_CLASSES / 16);
    gemm_bf16_wmma<FILTERS, NUM_CLASSES><<<(waves + 7) / 8, 256, 0, stream>>>(Hb, W2t, HW2, N_NODES);
  }
  // 7) logits = A @ HW2
  hipMemsetAsync(logits, 0, (size_t)N_NODES * NUM_CLASSES * 4, stream);
  {
    long thr = (long)N_EDGES * NUM_CLASSES;
    propagate_narrow<<<(int)((thr + 255) / 256), 256, 0, stream>>>(HW2, esrc, edst, ew, logits, N_EDGES);
  }
  // 8) out = softmax(logits)
  softmax16<<<(N_NODES + 255) / 256, 256, 0, stream>>>(logits, out, N_NODES);
}
